// BertWordEmbedder_26869315403947
// MI455X (gfx1250) — compile-verified, matrix operand order checked by reading, not run on hardware
//
#include <hip/hip_runtime.h>
#include <hip/hip_bf16.h>

// Problem constants (match the reference)
#define BB 64
#define TT 512
#define HH 768
#define WW 256
#define DD 256
#define MM (BB * WW)          // 16384 GEMM rows
#define KCH (HH / 32)         // 24 k-chunks of 32
#define MT (MM / 16)          // 1024 M tiles
#define NT (DD / 16)          // 16 N tiles

typedef __attribute__((ext_vector_type(16))) __bf16 v16bf;
typedef __attribute__((ext_vector_type(8)))  float  v8f;

// round-to-nearest-even f32 -> bf16
__device__ __forceinline__ unsigned short f2bf(float f) {
    union { float f; unsigned int u; } v; v.f = f;
    unsigned int u = v.u;
    u += 0x7FFFu + ((u >> 16) & 1u);
    return (unsigned short)(u >> 16);
}

// CDNA5 16-bit A/B fragment swizzle (CDNA5 ISA 7.12.2):
// within a 32-wide K chunk, for matrix row/col index rc (0..15):
//   kk in [0..7]   -> lane rc,    slot kk
//   kk in [8..15]  -> lane rc+16, slot kk-8
//   kk in [16..23] -> lane rc,    slot kk-8
//   kk in [24..31] -> lane rc+16, slot kk-16
__device__ __forceinline__ int frag_index(int tile, int chunk, int rc, int kk) {
    int group = kk >> 3;
    int lane  = rc + ((group & 1) << 4);
    int slot  = (kk & 7) + ((group >> 1) << 3);
    return (((tile * KCH + chunk) * 32) + lane) * 16 + slot;
}

// ---------------------------------------------------------------------------
// Kernel 1: token->word segment mean, output bf16 in WMMA A-fragment layout.
// One 256-thread block per (b, w). word_ids are sorted per row, so the word's
// tokens are a contiguous range found by binary search.
// ---------------------------------------------------------------------------
__global__ void __launch_bounds__(256)
pool_kernel(const float* __restrict__ hidden, const int* __restrict__ word_ids,
            unsigned short* __restrict__ wsA) {
    __shared__ int s_wid[TT];
    const int m   = blockIdx.x;          // row in [0, B*W)
    const int b   = m / WW;
    const int w   = m % WW;
    const int tid = threadIdx.x;

    // cache this sentence's word ids
    s_wid[tid]       = word_ids[b * TT + tid];
    s_wid[tid + 256] = word_ids[b * TT + tid + 256];
    __syncthreads();

    // lower_bound(w) and lower_bound(w+1)
    int lo = 0, hi = TT;
    while (lo < hi) { int mid = (lo + hi) >> 1; if (s_wid[mid] <  w) lo = mid + 1; else hi = mid; }
    const int start = lo;
    lo = start; hi = TT;
    while (lo < hi) { int mid = (lo + hi) >> 1; if (s_wid[mid] <= w) lo = mid + 1; else hi = mid; }
    const int end = lo;

    const int   cnt = end - start;
    const float inv = 1.0f / (float)max(cnt, 1);

    // accumulate 3 columns per thread (H = 768 = 3*256), coalesced over tid
    float s0 = 0.f, s1 = 0.f, s2 = 0.f;
    const float* hrow = hidden + (size_t)b * TT * HH;
    for (int t = start; t < end; ++t) {
        const float* p = hrow + (size_t)t * HH;
        s0 += p[tid];
        s1 += p[tid + 256];
        s2 += p[tid + 512];
    }
    s0 *= inv; s1 *= inv; s2 *= inv;

    // write swizzled bf16 A fragments for row m
    const int mtile = m >> 4, mrow = m & 15;
    float vals[3] = { s0, s1, s2 };
#pragma unroll
    for (int j = 0; j < 3; ++j) {
        int p  = tid + j * 256;         // h index
        int c  = p >> 5;                // k-chunk
        int kk = p & 31;
        wsA[frag_index(mtile, c, mrow, kk)] = f2bf(vals[j]);
    }
}

// ---------------------------------------------------------------------------
// Kernel 2: repack proj_w [H,D] f32 -> bf16 B fragments (mirrored layout, col=N)
// ---------------------------------------------------------------------------
__global__ void __launch_bounds__(256)
wconv_kernel(const float* __restrict__ projw, unsigned short* __restrict__ wsB) {
    int idx = blockIdx.x * 256 + threadIdx.x;   // over H*D
    int h = idx >> 8;                            // k
    int d = idx & 255;                           // n
    int ntile = d >> 4, ncol = d & 15;
    int c = h >> 5, kk = h & 31;
    wsB[frag_index(ntile, c, ncol, kk)] = f2bf(projw[idx]);
}

// ---------------------------------------------------------------------------
// Kernel 3: GEMM [M=16384, K=768] x [K, N=256] via v_wmma_f32_16x16x32_bf16.
// Block = 8 waves. Block stages the whole K-strip of B fragments for one
// N-tile in LDS (24 KB); each wave owns one 16-row M-tile.
// ---------------------------------------------------------------------------
__global__ void __launch_bounds__(256)
gemm_kernel(const unsigned short* __restrict__ wsA,
            const unsigned short* __restrict__ wsB,
            const float* __restrict__ bias,
            float* __restrict__ out) {
    __shared__ uint4 sBv[(KCH * 32 * 16) / 8];   // 24 KB of bf16 B fragments
    unsigned short* sB = (unsigned short*)sBv;

    const int tid   = threadIdx.x;
    const int lane  = tid & 31;
    const int wave  = tid >> 5;
    const int ntile = blockIdx.y;
    const int mtile = blockIdx.x * 8 + wave;

    // cooperative load of B strip: 1536 uint4, 6 per thread, fully coalesced
    const uint4* gB = (const uint4*)(wsB + (size_t)ntile * KCH * 32 * 16);
#pragma unroll
    for (int i = 0; i < 6; ++i)
        sBv[tid + i * 256] = gB[tid + i * 256];
    __syncthreads();

    v8f acc = {};
    const unsigned short* aBase =
        wsA + (((size_t)mtile * KCH) * 32 + lane) * 16;   // 32B-aligned per lane

#pragma unroll 4
    for (int c = 0; c < KCH; ++c) {
        v16bf a = *(const v16bf*)(aBase + (size_t)c * 32 * 16);
        v16bf bm = *(const v16bf*)(sB + (c * 32 + lane) * 16);
        acc = __builtin_amdgcn_wmma_f32_16x16x32_bf16(
            /*neg_a=*/false, a, /*neg_b=*/false, bm,
            /*c_mod=*/(short)0, acc, /*reuse_a=*/false, /*reuse_b=*/false);
    }

    // C/D layout: VGPR r -> (M = r + (lane>=16 ? 8 : 0), N = lane & 15)
    const int col   = ntile * 16 + (lane & 15);
    const float bv  = bias[col];
    const int rbase = mtile * 16 + ((lane >> 4) << 3);
#pragma unroll
    for (int r = 0; r < 8; ++r)
        out[(size_t)(rbase + r) * DD + col] = acc[r] + bv;
}

// ---------------------------------------------------------------------------
extern "C" void kernel_launch(void* const* d_in, const int* in_sizes, int n_in,
                              void* d_out, int out_size, void* d_ws, size_t ws_size,
                              hipStream_t stream) {
    const float* hidden   = (const float*)d_in[0];  // [B,T,H] f32
    const int*   word_ids = (const int*)d_in[1];    // [B,T]   i32
    const float* proj_w   = (const float*)d_in[2];  // [H,D]   f32
    const float* proj_b   = (const float*)d_in[3];  // [D]     f32
    float*       out      = (float*)d_out;          // [B,W,D] f32

    unsigned short* wsA = (unsigned short*)d_ws;          // M*H bf16 = 24 MB
    unsigned short* wsB = wsA + (size_t)MM * HH;          // H*D bf16 = 384 KB

    pool_kernel<<<MM, 256, 0, stream>>>(hidden, word_ids, wsA);
    wconv_kernel<<<(HH * DD) / 256, 256, 0, stream>>>(proj_w, wsB);
    gemm_kernel<<<dim3(MT / 8, NT), 256, 0, stream>>>(wsA, wsB, proj_b, out);
}
